// BiLSTMPhishDetector_86114094285233
// MI455X (gfx1250) — compile-verified, matrix operand order checked by reading, not run on hardware
//
#include <hip/hip_runtime.h>
#include <hip/hip_fp16.h>

#define B_SZ 2048
#define L_SZ 256

typedef _Float16 h8_t  __attribute__((ext_vector_type(8)));
typedef _Float16 h16_t __attribute__((ext_vector_type(16)));
typedef float    f8_t  __attribute__((ext_vector_type(8)));

__device__ __forceinline__ float fsigmoid(float x) { return 1.0f / (1.0f + __expf(-x)); }
__device__ __forceinline__ float ftanh_fast(float x) {
  float t = __expf(-2.0f * fabsf(x));
  float r = (1.0f - t) / (1.0f + t);
  return copysignf(r, x);
}

// Load one 16x32 f16 WMMA fragment (A or B operand) from a row-major LDS matrix.
// Per ISA layout: lane<16 -> K base 0, lane>=16 -> K base 8; halves 0..7 are
// K=base..base+7, halves 8..15 are K=base+16..base+23 (two contiguous 16B loads).
__device__ __forceinline__ h16_t load_frag16(const _Float16* row, int kchunk, int lane) {
  const int base = kchunk * 32 + ((lane & 16) ? 8 : 0);
  h8_t lo = *(const h8_t*)(row + base);
  h8_t hi = *(const h8_t*)(row + base + 16);
  h16_t r;
#pragma unroll
  for (int j = 0; j < 8; ++j) { r[j] = lo[j]; r[j + 8] = hi[j]; }
  return r;
}

// One direction of one BiLSTM layer. Block = 16 batch rows, 128 threads (4 waves).
// K = input_dim + 64 (layer0: 64+64=128, layer1: 128+64=192).
// Wave w handles hidden cols [16w,16w+16) for all 4 gates -> in-register gate fusion.
template <int K, int LAYER>
__global__ void __launch_bounds__(128) lstm_dir_kernel(
    const int* __restrict__ x,          // [B,L]           (layer 0 only)
    const float* __restrict__ emb,      // [128,64]        (layer 0 only)
    const _Float16* __restrict__ yin,   // [L,B,128] f16   (layer 1 only)
    const float* __restrict__ Wih,      // [2,256,K-64]
    const float* __restrict__ Whh,      // [2,256,64]
    const float* __restrict__ bias,     // [2,2,256]
    _Float16* __restrict__ yout,        // [L,B,128] f16   (layer 0 only)
    float* __restrict__ hfin) {         // [2,B,64] f32    (layer 1 only)
  constexpr int D   = K - 64;   // input feature dim
  constexpr int LDA = K + 8;    // padded LDS row stride (halves) to spread banks
  const int dir  = blockIdx.y;
  const int b0   = blockIdx.x * 16;
  const int tid  = threadIdx.x;
  const int wave = tid >> 5;
  const int lane = tid & 31;

  extern __shared__ char smem_raw[];
  _Float16* sW   = (_Float16*)smem_raw;                 // [256][LDA] combined [Wih|Whh] f16
  _Float16* sA   = sW + 256 * LDA;                      // [16][LDA]  [x_t | h_{t-1}] f16
  float*    sB   = (float*)(sA + 16 * LDA);             // [256] bih+bhh
  _Float16* sEmb = (_Float16*)(sB + 256);               // [128][64] f16 (layer 0 only)

  // ---- preload combined weight matrix (f32 -> f16) and bias ----
  const float* wih = Wih + (size_t)dir * 256 * D;
  const float* whh = Whh + (size_t)dir * 256 * 64;
  for (int idx = tid; idx < 256 * K; idx += 128) {
    const int n = idx / K, k = idx % K;
    const float v = (k < D) ? wih[n * D + k] : whh[n * 64 + (k - D)];
    sW[n * LDA + k] = (_Float16)v;
  }
  for (int n = tid; n < 256; n += 128)
    sB[n] = bias[dir * 512 + n] + bias[dir * 512 + 256 + n];
  if constexpr (LAYER == 0) {
    for (int idx = tid; idx < 128 * 64; idx += 128) sEmb[idx] = (_Float16)emb[idx];
  }
  // zero the h-region of sA (h_0 = 0)
  for (int idx = tid; idx < 16 * 64; idx += 128)
    sA[(idx >> 6) * LDA + D + (idx & 63)] = (_Float16)0.0f;
  __syncthreads();

  const int nrow  = lane & 15;             // N column within a 16x16 tile
  const int mbase = (lane & 16) ? 8 : 0;   // C-layout: VGPR r holds M = r + mbase
  const int hcol  = wave * 16 + nrow;      // hidden column owned by this lane

  const float bi = sB[0 * 64 + hcol];
  const float bf = sB[1 * 64 + hcol];
  const float bg = sB[2 * 64 + hcol];
  const float bo = sB[3 * 64 + hcol];

  // ---- hoist B fragments (weights) into registers: const over the time loop ----
  h16_t bfrag[4][K / 32];
#pragma unroll
  for (int g = 0; g < 4; ++g) {
    const _Float16* wrow = sW + (g * 64 + hcol) * LDA;  // B row index = output column
#pragma unroll
    for (int kc = 0; kc < K / 32; ++kc) bfrag[g][kc] = load_frag16(wrow, kc, lane);
  }

  f8_t c_st;
#pragma unroll
  for (int r = 0; r < 8; ++r) c_st[r] = 0.0f;

  for (int s = 0; s < L_SZ; ++s) {
    const int t = dir ? (L_SZ - 1 - s) : s;

    // ---- stage x_t into sA[:, 0:D) (128 threads, 16 rows) ----
    if constexpr (LAYER == 0) {
      const int row = tid >> 3, c8 = (tid & 7) * 8;
      const int tok = x[(b0 + row) * L_SZ + t];
      *(h8_t*)(sA + row * LDA + c8) = *(const h8_t*)(sEmb + tok * 64 + c8);
    } else {
      const int row = tid >> 3, c16 = (tid & 7) * 16;
      const _Float16* src = yin + ((size_t)t * B_SZ + (b0 + row)) * 128 + c16;
      *(h8_t*)(sA + row * LDA + c16)     = *(const h8_t*)(src);
      *(h8_t*)(sA + row * LDA + c16 + 8) = *(const h8_t*)(src + 8);
      if (s < L_SZ - 1) {  // prefetch next timestep's activations
        const int tn = dir ? (t - 1) : (t + 1);
        __builtin_prefetch(yin + ((size_t)tn * B_SZ + (b0 + row)) * 128 + c16, 0, 0);
      }
    }
    __syncthreads();  // staged x_t + previous step's h writes visible

    // ---- load A fragments [x_t | h_{t-1}] ----
    h16_t afrag[K / 32];
    const _Float16* arow = sA + nrow * LDA;
#pragma unroll
    for (int kc = 0; kc < K / 32; ++kc) afrag[kc] = load_frag16(arow, kc, lane);
    __syncthreads();  // all waves done reading sA before h/x overwrite

    // ---- gates = [x|h] @ [Wih|Whh]^T + b  (4 tiles x K/32 chained WMMAs) ----
    f8_t acc_i, acc_f, acc_g, acc_o;
#pragma unroll
    for (int r = 0; r < 8; ++r) { acc_i[r] = bi; acc_f[r] = bf; acc_g[r] = bg; acc_o[r] = bo; }
#pragma unroll
    for (int kc = 0; kc < K / 32; ++kc) {
      acc_i = __builtin_amdgcn_wmma_f32_16x16x32_f16(false, afrag[kc], false, bfrag[0][kc],
                                                     (short)0, acc_i, false, false);
      acc_f = __builtin_amdgcn_wmma_f32_16x16x32_f16(false, afrag[kc], false, bfrag[1][kc],
                                                     (short)0, acc_f, false, false);
      acc_g = __builtin_amdgcn_wmma_f32_16x16x32_f16(false, afrag[kc], false, bfrag[2][kc],
                                                     (short)0, acc_g, false, false);
      acc_o = __builtin_amdgcn_wmma_f32_16x16x32_f16(false, afrag[kc], false, bfrag[3][kc],
                                                     (short)0, acc_o, false, false);
    }

    // ---- LSTM cell update (elementwise, all four gates resident per lane) ----
    _Float16 hh[8];
    float    hf32[8];
#pragma unroll
    for (int r = 0; r < 8; ++r) {
      const float cv = fsigmoid(acc_f[r]) * c_st[r] + fsigmoid(acc_i[r]) * ftanh_fast(acc_g[r]);
      c_st[r] = cv;
      const float hv = fsigmoid(acc_o[r]) * ftanh_fast(cv);
      hf32[r] = hv;
      hh[r]   = (_Float16)hv;
    }

    // ---- feed h back through LDS; emit outputs ----
#pragma unroll
    for (int r = 0; r < 8; ++r) {
      const int M = r + mbase;
      sA[M * LDA + D + hcol] = hh[r];
      if constexpr (LAYER == 0)
        yout[((size_t)t * B_SZ + (b0 + M)) * 128 + dir * 64 + hcol] = hh[r];
    }
    if constexpr (LAYER == 1) {
      if (s == L_SZ - 1) {
#pragma unroll
        for (int r = 0; r < 8; ++r) {
          const int M = r + mbase;
          hfin[((size_t)dir * B_SZ + (b0 + M)) * 64 + hcol] = hf32[r];
        }
      }
    }
  }
}

// out[b] = W2 @ relu(W1 @ [hf|hb] + b1) + b2   (tiny: plain VALU)
__global__ void __launch_bounds__(64) head_kernel(
    const float* __restrict__ hfin, const float* __restrict__ W1,
    const float* __restrict__ bf1, const float* __restrict__ W2,
    const float* __restrict__ bf2, float* __restrict__ out) {
  const int b = blockIdx.x;
  const int j = threadIdx.x;  // 0..63
  __shared__ float sh[128];
  __shared__ float red[64];
  sh[j]      = hfin[(size_t)b * 64 + j];                      // hf
  sh[64 + j] = hfin[(size_t)(B_SZ + b) * 64 + j];             // hb
  __syncthreads();
  float acc = bf1[j];
#pragma unroll 4
  for (int k = 0; k < 128; ++k) acc += W1[j * 128 + k] * sh[k];
  red[j] = fmaxf(acc, 0.0f) * W2[j];
  __syncthreads();
  if (j == 0) {
    float s = 0.0f;
    for (int k = 0; k < 64; ++k) s += red[k];
    out[b] = s + bf2[0];
  }
}

extern "C" void kernel_launch(void* const* d_in, const int* in_sizes, int n_in,
                              void* d_out, int out_size, void* d_ws, size_t ws_size,
                              hipStream_t stream) {
  const int*   x    = (const int*)d_in[0];
  const float* emb  = (const float*)d_in[1];
  const float* Wih0 = (const float*)d_in[2];
  const float* Whh0 = (const float*)d_in[3];
  const float* b0   = (const float*)d_in[4];
  const float* Wih1 = (const float*)d_in[5];
  const float* Whh1 = (const float*)d_in[6];
  const float* b1   = (const float*)d_in[7];
  const float* W1   = (const float*)d_in[8];
  const float* bf1  = (const float*)d_in[9];
  const float* W2   = (const float*)d_in[10];
  const float* bf2  = (const float*)d_in[11];
  float* out = (float*)d_out;

  // workspace: y0 [L,B,128] f16 (128MB), then hfin [2,B,64] f32 (1MB)
  _Float16* y0   = (_Float16*)d_ws;
  float*    hfin = (float*)((char*)d_ws + (size_t)L_SZ * B_SZ * 128 * sizeof(_Float16));

  constexpr size_t smem0 = (size_t)(256 + 16) * (128 + 8) * 2 + 256 * 4 + 128 * 64 * 2;
  constexpr size_t smem1 = (size_t)(256 + 16) * (192 + 8) * 2 + 256 * 4;
  (void)hipFuncSetAttribute(reinterpret_cast<const void*>(lstm_dir_kernel<128, 0>),
                            hipFuncAttributeMaxDynamicSharedMemorySize, (int)smem0);
  (void)hipFuncSetAttribute(reinterpret_cast<const void*>(lstm_dir_kernel<192, 1>),
                            hipFuncAttributeMaxDynamicSharedMemorySize, (int)smem1);

  dim3 grid(B_SZ / 16, 2);  // 128 batch tiles x 2 directions
  lstm_dir_kernel<128, 0><<<grid, 128, smem0, stream>>>(x, emb, nullptr, Wih0, Whh0, b0, y0,
                                                        nullptr);
  lstm_dir_kernel<192, 1><<<grid, 128, smem1, stream>>>(nullptr, nullptr, y0, Wih1, Whh1, b1,
                                                        nullptr, hfin);
  head_kernel<<<B_SZ, 64, 0, stream>>>(hfin, W1, bf1, W2, bf2, out);
}
